// GNN_41798621724824
// MI455X (gfx1250) — matrix operand order repaired and checked by reading, not compiled, not run on hardware
//
#include <hip/hip_runtime.h>
#include <hip/hip_bf16.h>

// ---------------- problem constants (match reference) ----------------
#define BGR   64           // graphs
#define NPG   1024         // nodes per graph
#define NTOT  (BGR*NPG)    // 65536
#define KNN   5
#define FEAT  3
#define POS   3
#define HID   64
#define NLAY  5

#define CEILDIV(a,b) (((a)+(b)-1)/(b))

typedef __bf16 v16bf __attribute__((ext_vector_type(16)));
typedef float  v8f   __attribute__((ext_vector_type(8)));

union Frag32B { uint4 u[2]; v16bf v; };

__device__ __forceinline__ float bf2f(unsigned short h) {
    return __uint_as_float(((unsigned)h) << 16);
}
__device__ __forceinline__ unsigned short f2bf(float f) {
    unsigned u = __float_as_uint(f);
    u += 0x7FFFu + ((u >> 16) & 1u);           // round-to-nearest-even
    return (unsigned short)(u >> 16);
}

// --------- CDNA5 async global->LDS copy (guarded; 16B per lane) ---------
typedef int v4i_t __attribute__((vector_size(16)));
typedef __attribute__((address_space(1))) v4i_t GV4;
typedef __attribute__((address_space(3))) v4i_t LV4;

__device__ __forceinline__ void async_copy16(const void* g, void* l) {
#if __has_builtin(__builtin_amdgcn_global_load_async_to_lds_b128)
    __builtin_amdgcn_global_load_async_to_lds_b128((GV4*)g, (LV4*)l, 0, 0);
#else
    *(uint4*)l = *(const uint4*)g;             // fallback: load + ds_store
#endif
}
__device__ __forceinline__ void wait_async0() {
#if __has_builtin(__builtin_amdgcn_s_wait_asynccnt)
    __builtin_amdgcn_s_wait_asynccnt(0);
#else
    asm volatile("s_wait_asynccnt 0x0" ::: "memory");
#endif
}

// ---------------------------------------------------------------------
// kNN: one block per graph; positions staged in LDS; per-thread top-5.
// ---------------------------------------------------------------------
__global__ __launch_bounds__(256)
void knn_kernel(const float* __restrict__ pos, int* __restrict__ src) {
    __shared__ float px[NPG], py[NPG], pz[NPG];
    const int b = blockIdx.x;
    const float* P = pos + (size_t)b * NPG * POS;
    for (int i = threadIdx.x; i < NPG; i += blockDim.x) {
        px[i] = P[i*3+0]; py[i] = P[i*3+1]; pz[i] = P[i*3+2];
    }
    __syncthreads();
    for (int q = threadIdx.x; q < NPG; q += blockDim.x) {
        const float qx = px[q], qy = py[q], qz = pz[q];
        float bd[KNN]; int bi[KNN];
        #pragma unroll
        for (int k = 0; k < KNN; ++k) { bd[k] = 3.4e38f; bi[k] = 0; }
        for (int j = 0; j < NPG; ++j) {
            const float dx = px[j]-qx, dy = py[j]-qy, dz = pz[j]-qz;
            const float d2 = dx*dx + dy*dy + dz*dz;
            if (d2 < bd[KNN-1]) {
                int p = KNN-1;
                while (p > 0 && bd[p-1] > d2) { bd[p]=bd[p-1]; bi[p]=bi[p-1]; --p; }
                bd[p] = d2; bi[p] = j;
            }
        }
        const int base = (b*NPG + q) * KNN;
        #pragma unroll
        for (int k = 0; k < KNN; ++k) src[base + k] = b*NPG + bi[k];
    }
}

// ---------------------------------------------------------------------
// Stage-0 embed: h = [x,pos] @ We(6x64) + be   (din=6, not WMMA-shaped)
// ---------------------------------------------------------------------
__global__ __launch_bounds__(256)
void embed0_kernel(const float* __restrict__ x, const float* __restrict__ pos,
                   const float* __restrict__ We, const float* __restrict__ be,
                   float* __restrict__ H) {
    const long idx = (long)blockIdx.x * blockDim.x + threadIdx.x;
    if (idx >= (long)NTOT * HID) return;
    const int i = (int)(idx / HID), c = (int)(idx % HID);
    float in[6];
    in[0]=x[i*3+0]; in[1]=x[i*3+1]; in[2]=x[i*3+2];
    in[3]=pos[i*3+0]; in[4]=pos[i*3+1]; in[5]=pos[i*3+2];
    float acc = be[c];
    #pragma unroll
    for (int f = 0; f < 6; ++f) acc += in[f] * We[f*HID + c];
    H[idx] = acc;
}

// ---------------------------------------------------------------------
// Degrees + edge weights (normalized Laplacian term)
// ---------------------------------------------------------------------
__global__ __launch_bounds__(256)
void fill0_kernel(float* __restrict__ p, int n) {
    const int i = blockIdx.x * blockDim.x + threadIdx.x;
    if (i < n) p[i] = 0.0f;
}

__global__ __launch_bounds__(256)
void deg_kernel(const int* __restrict__ src, int shift, float* __restrict__ deg, int nE) {
    const int e = blockIdx.x * blockDim.x + threadIdx.x;
    if (e >= nE) return;
    const int s = src[e] >> shift;
    const int d = (e / KNN) >> shift;
    if (s != d) atomicAdd(&deg[s], 1.0f);
}

__global__ __launch_bounds__(256)
void edgew_kernel(const int* __restrict__ src, const float* __restrict__ deg,
                  float* __restrict__ w, int shift, int nE) {
    const int e = blockIdx.x * blockDim.x + threadIdx.x;
    if (e >= nE) return;
    const int s = src[e] >> shift;
    const int d = (e / KNN) >> shift;
    if (s == d) { w[e] = 0.0f; return; }
    const float ds = deg[s], dd = deg[d];
    const float is = (ds > 0.f) ? rsqrtf(fmaxf(ds, 1e-12f)) : 0.f;
    const float id = (dd > 0.f) ? rsqrtf(fmaxf(dd, 1e-12f)) : 0.f;
    w[e] = -is * id;
}

// ---------------------------------------------------------------------
// BatchNorm stats: one block per channel (column), 256-thread reduction.
// ---------------------------------------------------------------------
__global__ __launch_bounds__(256)
void bn_stats_kernel(const float* __restrict__ X, int n, int d,
                     float* __restrict__ mean, float* __restrict__ rstd) {
    const int c = blockIdx.x;
    float s = 0.f, s2 = 0.f;
    for (int i = threadIdx.x; i < n; i += 256) {
        const float v = X[(size_t)i * d + c];
        s += v; s2 += v * v;
    }
    __shared__ float sh[256], sh2[256];
    sh[threadIdx.x] = s; sh2[threadIdx.x] = s2;
    __syncthreads();
    for (int st = 128; st > 0; st >>= 1) {
        if (threadIdx.x < st) { sh[threadIdx.x] += sh[threadIdx.x+st]; sh2[threadIdx.x] += sh2[threadIdx.x+st]; }
        __syncthreads();
    }
    if (threadIdx.x == 0) {
        const float m = sh[0] / (float)n;
        const float v = sh2[0] / (float)n - m * m;
        mean[c] = m;
        rstd[c] = rsqrtf(v + 1e-5f);
    }
}

// relu(bn(X)) -> bf16
__global__ __launch_bounds__(256)
void bnrelu_bf16_kernel(const float* __restrict__ X, const float* __restrict__ mean,
                        const float* __restrict__ rstd, const float* __restrict__ g,
                        const float* __restrict__ b, unsigned short* __restrict__ Y,
                        long total, int d) {
    const long idx = (long)blockIdx.x * blockDim.x + threadIdx.x;
    if (idx >= total) return;
    const int c = (int)(idx % d);
    float v = (X[idx] - mean[c]) * rstd[c] * g[c] + b[c];
    v = fmaxf(v, 0.0f);
    Y[idx] = f2bf(v);
}

// H = relu(bn(T)) + H   (residual, in place)
__global__ __launch_bounds__(256)
void bnrelu_add_kernel(const float* __restrict__ T, const float* __restrict__ mean,
                       const float* __restrict__ rstd, const float* __restrict__ g,
                       const float* __restrict__ b, float* __restrict__ H,
                       long total, int d) {
    const long idx = (long)blockIdx.x * blockDim.x + threadIdx.x;
    if (idx >= total) return;
    const int c = (int)(idx % d);
    float v = (T[idx] - mean[c]) * rstd[c] * g[c] + b[c];
    H[idx] = fmaxf(v, 0.0f) + H[idx];
}

// relu(bn(X)) -> f32 (head path, in-place safe)
__global__ __launch_bounds__(256)
void bnrelu_f32_kernel(const float* __restrict__ X, const float* __restrict__ mean,
                       const float* __restrict__ rstd, const float* __restrict__ g,
                       const float* __restrict__ b, float* __restrict__ Y,
                       long total, int d) {
    const long idx = (long)blockIdx.x * blockDim.x + threadIdx.x;
    if (idx >= total) return;
    const int c = (int)(idx % d);
    float v = (X[idx] - mean[c]) * rstd[c] * g[c] + b[c];
    Y[idx] = fmaxf(v, 0.0f);
}

// ---------------------------------------------------------------------
// Cheb aggregation: edges for coarse node j are contiguous -> gather.
// ---------------------------------------------------------------------
__global__ void cheb_kernel(const unsigned short* __restrict__ A,
                            const int* __restrict__ src, const float* __restrict__ w,
                            int shift, unsigned short* __restrict__ T1, int d) {
    const int node = blockIdx.x;
    const int c = threadIdx.x;
    const int ne = KNN << shift;                 // <= 20
    __shared__ int   ssrc[32];
    __shared__ float swt[32];
    if (c < ne) {
        const int e = node * ne + c;
        ssrc[c] = src[e] >> shift;
        swt[c]  = w[e];
    }
    __syncthreads();
    float acc = 0.f;
    for (int t = 0; t < ne; ++t)
        acc += swt[t] * bf2f(A[(size_t)ssrc[t] * d + c]);
    T1[(size_t)node * d + c] = f2bf(acc);
}

// W[din][dout] f32 -> Wt[dout][din] bf16
__global__ __launch_bounds__(256)
void transpose_cast_kernel(const float* __restrict__ W, unsigned short* __restrict__ Wt,
                           int din, int dout) {
    const int idx = blockIdx.x * blockDim.x + threadIdx.x;
    if (idx >= din * dout) return;
    const int k = idx / dout, j = idx % dout;
    Wt[(size_t)j * din + k] = f2bf(W[idx]);
}

__global__ __launch_bounds__(256)
void cast_bf16_kernel(const float* __restrict__ X, unsigned short* __restrict__ Y, long total) {
    const long idx = (long)blockIdx.x * blockDim.x + threadIdx.x;
    if (idx < total) Y[idx] = f2bf(X[idx]);
}

// ---------------------------------------------------------------------
// WMMA GEMM (wave32): D[n,dout] = A1@W1t (+ A2@W2t) + bias
//   - A-fragments register-cached across the whole j-loop (K unrolled).
//   - B j-tiles staged in LDS once per block via async global->LDS copies,
//     double-buffered; rows padded (+8 halves) to avoid bank conflicts.
//   - n % 128 == 0 always: every wave has full EXEC for WMMA.
// ---------------------------------------------------------------------
template <int DIN, bool HAS2>
__global__ __launch_bounds__(256)
void wmma_gemm_kernel(const unsigned short* __restrict__ A1,
                      const unsigned short* __restrict__ A2,
                      const unsigned short* __restrict__ W1t,
                      const unsigned short* __restrict__ W2t,
                      const float* __restrict__ bias,
                      float* __restrict__ D, int dout) {
    constexpr int KF   = DIN / 32;          // K fragments per tile row
    constexpr int LROW = DIN + 8;           // padded LDS row (halves)
    constexpr int NMAT = HAS2 ? 2 : 1;
    constexpr int CPR  = DIN / 8;           // 16B chunks per row
    constexpr int NCH  = 16 * CPR;          // chunks per matrix tile

    __shared__ __align__(16) unsigned short sB[2][NMAT * 16 * LROW];

    const int wave = threadIdx.x >> 5;
    const int lane = threadIdx.x & 31;
    const int half = lane >> 4;             // 0: lanes 0-15, 1: lanes 16-31
    const int m    = lane & 15;
    const int mbase = (blockIdx.x * 8 + wave) * 16;

    // ---- register-cache the wave's A strip (16 rows x DIN) ----
    Frag32B afr1[KF], afr2[KF];
    {
        const unsigned short* arow1 = A1 + (size_t)(mbase + m) * DIN;
        #pragma unroll
        for (int kf = 0; kf < KF; ++kf) {
            const int kb = kf * 32;
            afr1[kf].u[0] = *(const uint4*)(arow1 + kb + half * 8);
            afr1[kf].u[1] = *(const uint4*)(arow1 + kb + 16 + half * 8);
        }
        if (HAS2) {
            const unsigned short* arow2 = A2 + (size_t)(mbase + m) * DIN;
            #pragma unroll
            for (int kf = 0; kf < KF; ++kf) {
                const int kb = kf * 32;
                afr2[kf].u[0] = *(const uint4*)(arow2 + kb + half * 8);
                afr2[kf].u[1] = *(const uint4*)(arow2 + kb + 16 + half * 8);
            }
        }
    }

    // ---- B tile staging: cooperative async copy of Wt[jb..jb+16)[:] ----
    auto stage = [&](int buf, int jb) {
        const int total = NMAT * NCH;
        for (int t = threadIdx.x; t < total; t += 256) {
            const int mat = t / NCH;
            const int c   = t % NCH;
            const int r   = c / CPR;
            const int cc  = c % CPR;
            const unsigned short* sp =
                (mat ? W2t : W1t) + (size_t)(jb + r) * DIN + cc * 8;
            unsigned short* dp_ = &sB[buf][mat * 16 * LROW + r * LROW + cc * 8];
            async_copy16(sp, dp_);
        }
    };

    const int nj = dout >> 4;
    stage(0, 0);

    for (int i = 0; i < nj; ++i) {
        const int cur = i & 1;
        const int jb  = i << 4;
        wait_async0();                       // this wave's staging complete
        __syncthreads();                     // whole block's staging complete
        if (i + 1 < nj) stage(cur ^ 1, jb + 16);

        v8f acc;
        const float bv = bias[jb + m];       // C layout: all rows share col N=m
        #pragma unroll
        for (int r = 0; r < 8; ++r) acc[r] = bv;

        const unsigned short* b1 = &sB[cur][m * LROW];
        #pragma unroll
        for (int kf = 0; kf < KF; ++kf) {
            const int kb = kf * 32;
            Frag32B b;
            b.u[0] = *(const uint4*)(b1 + kb + half * 8);
            b.u[1] = *(const uint4*)(b1 + kb + 16 + half * 8);
            acc = __builtin_amdgcn_wmma_f32_16x16x32_bf16(
                false, afr1[kf].v, false, b.v, (short)0, acc, false, false);
            if (HAS2) {
                const unsigned short* b2 = b1 + 16 * LROW;
                Frag32B bb;
                bb.u[0] = *(const uint4*)(b2 + kb + half * 8);
                bb.u[1] = *(const uint4*)(b2 + kb + 16 + half * 8);
                acc = __builtin_amdgcn_wmma_f32_16x16x32_bf16(
                    false, afr2[kf].v, false, bb.v, (short)0, acc, false, false);
            }
        }
        // f32 C/D layout: VGPR r -> row = mbase + half*8 + r, col = jb + m
        float* dp = D + (size_t)(mbase + half * 8) * dout + jb + m;
        #pragma unroll
        for (int r = 0; r < 8; ++r) dp[(size_t)r * dout] = acc[r];
    }
}

// pooled[j] = 0.5*(H[2j] + H[2j+1])
__global__ __launch_bounds__(256)
void pool_kernel(const float* __restrict__ H, float* __restrict__ Hp, long total, int d) {
    const long idx = (long)blockIdx.x * blockDim.x + threadIdx.x;
    if (idx >= total) return;
    const long j = idx / d; const int c = (int)(idx % d);
    Hp[idx] = 0.5f * (H[(size_t)(2*j) * d + c] + H[(size_t)(2*j + 1) * d + c]);
}

// graph mean: 64 blocks x 256 channels, 128 nodes per graph (n=8192, d=256)
__global__ __launch_bounds__(256)
void graph_mean_kernel(const float* __restrict__ H, float* __restrict__ G) {
    const int b = blockIdx.x, c = threadIdx.x;
    float acc = 0.f;
    for (int i = 0; i < 128; ++i)
        acc += H[(size_t)(b * 128 + i) * 256 + c];
    G[b * 256 + c] = acc * (1.0f / 128.0f);
}

// tiny dense head GEMM (64 rows; scalar path is fine here)
__global__ __launch_bounds__(128)
void small_gemm_kernel(const float* __restrict__ X, const float* __restrict__ W,
                       const float* __restrict__ bias, float* __restrict__ Y,
                       int n, int din, int dout) {
    const int idx = blockIdx.x * blockDim.x + threadIdx.x;
    if (idx >= n * dout) return;
    const int i = idx / dout, j = idx % dout;
    float acc = bias[j];
    for (int k = 0; k < din; ++k) acc += X[i * din + k] * W[k * dout + j];
    Y[idx] = acc;
}

__global__ __launch_bounds__(64)
void softmax_kernel(const float* __restrict__ X, float* __restrict__ Y, int d) {
    const int row = blockIdx.x, c = threadIdx.x;
    __shared__ float sh[64];
    const float v = X[row * d + c];
    sh[c] = v; __syncthreads();
    for (int st = 32; st > 0; st >>= 1) {
        if (c < st) sh[c] = fmaxf(sh[c], sh[c + st]);
        __syncthreads();
    }
    const float mx = sh[0]; __syncthreads();
    const float e = __expf(v - mx);
    sh[c] = e; __syncthreads();
    for (int st = 32; st > 0; st >>= 1) {
        if (c < st) sh[c] += sh[c + st];
        __syncthreads();
    }
    Y[row * d + c] = e / sh[0];
}

// ---------------------------------------------------------------------
// Host orchestration
// ---------------------------------------------------------------------
static void launch_gemm(const unsigned short* A1, const unsigned short* A2,
                        const unsigned short* W1t, const unsigned short* W2t,
                        const float* bias, float* D, int n, int din, int dout,
                        hipStream_t stream) {
    const int grid = n / 128;
    if (A2) {
        if (din == 64)       wmma_gemm_kernel< 64, true><<<grid, 256, 0, stream>>>(A1, A2, W1t, W2t, bias, D, dout);
        else if (din == 128) wmma_gemm_kernel<128, true><<<grid, 256, 0, stream>>>(A1, A2, W1t, W2t, bias, D, dout);
        else                 wmma_gemm_kernel<256, true><<<grid, 256, 0, stream>>>(A1, A2, W1t, W2t, bias, D, dout);
    } else {
        if (din == 64)       wmma_gemm_kernel< 64, false><<<grid, 256, 0, stream>>>(A1, nullptr, W1t, nullptr, bias, D, dout);
        else if (din == 128) wmma_gemm_kernel<128, false><<<grid, 256, 0, stream>>>(A1, nullptr, W1t, nullptr, bias, D, dout);
        else                 wmma_gemm_kernel<256, false><<<grid, 256, 0, stream>>>(A1, nullptr, W1t, nullptr, bias, D, dout);
    }
}

struct LayerP { const float *W0,*W1,*bn_b,*bn_g,*cb,*m1b,*m1g,*m2b,*m2g,*mW,*mb; };
struct StageP { const float *We,*be; LayerP L[NLAY]; };

extern "C" void kernel_launch(void* const* d_in, const int* in_sizes, int n_in,
                              void* d_out, int out_size, void* d_ws, size_t ws_size,
                              hipStream_t stream) {
    (void)in_sizes; (void)n_in; (void)out_size; (void)ws_size;
    // ---- unpack inputs (jax pytree flatten order: dict keys sorted) ----
    int ix = 0;
    const float* x    = (const float*)d_in[ix++];
    const float* pos  = (const float*)d_in[ix++];
    const int*   batch= (const int*)d_in[ix++]; (void)batch;
    // params: 'head' < 'stages'; head keys: W0,W1,b0,b1,bn0b,bn0g,bn1b,bn1g,bn2b,bn2g
    const float* hW0  = (const float*)d_in[ix++];
    const float* hW1  = (const float*)d_in[ix++];
    const float* hb0  = (const float*)d_in[ix++];
    const float* hb1  = (const float*)d_in[ix++];
    const float* hbn0b= (const float*)d_in[ix++];
    const float* hbn0g= (const float*)d_in[ix++];
    const float* hbn1b= (const float*)d_in[ix++];
    const float* hbn1g= (const float*)d_in[ix++];
    const float* hbn2b= (const float*)d_in[ix++];
    const float* hbn2g= (const float*)d_in[ix++];
    // stages: keys sorted: We, be, layers; layer keys sorted:
    // W0,W1,bn_b,bn_g,cb,m1b,m1g,m2b,m2g,mW,mb
    StageP st[3];
    for (int s = 0; s < 3; ++s) {
        st[s].We = (const float*)d_in[ix++];
        st[s].be = (const float*)d_in[ix++];
        for (int l = 0; l < NLAY; ++l) {
            LayerP& L = st[s].L[l];
            L.W0  =(const float*)d_in[ix++]; L.W1 =(const float*)d_in[ix++];
            L.bn_b=(const float*)d_in[ix++]; L.bn_g=(const float*)d_in[ix++];
            L.cb  =(const float*)d_in[ix++];
            L.m1b =(const float*)d_in[ix++]; L.m1g=(const float*)d_in[ix++];
            L.m2b =(const float*)d_in[ix++]; L.m2g=(const float*)d_in[ix++];
            L.mW  =(const float*)d_in[ix++]; L.mb =(const float*)d_in[ix++];
        }
    }
    static const int DIMS[3] = { HID, 2*HID, 4*HID };

    // ---- workspace layout ----
    char* wp = (char*)d_ws;
    auto alloc = [&](size_t bytes) -> void* {
        void* p = (void*)wp; wp += (bytes + 255) & ~(size_t)255; return p;
    };
    const size_t ACT = (size_t)NTOT * HID;                 // 4M elements (same every stage)
    float*          Hb  = (float*)alloc(ACT * 4);
    float*          Tb  = (float*)alloc(ACT * 4);
    unsigned short* A1  = (unsigned short*)alloc(ACT * 2);
    unsigned short* A2  = (unsigned short*)alloc(ACT * 2);
    unsigned short* WT1 = (unsigned short*)alloc(256 * 256 * 2);
    unsigned short* WT2 = (unsigned short*)alloc(256 * 256 * 2);
    float*          bnm = (float*)alloc(256 * 4);
    float*          bnr = (float*)alloc(256 * 4);
    int*            src = (int*)alloc((size_t)NTOT * KNN * 4);
    float*          deg = (float*)alloc((size_t)NTOT * 4);
    float*          wE  = (float*)alloc((size_t)NTOT * KNN * 4);
    float*          G0  = (float*)alloc(64 * 256 * 4);
    float*          G1  = (float*)alloc(64 * 128 * 4);
    float*          G2  = (float*)alloc(64 * 64 * 4);

    const int nE = NTOT * KNN;

    // ---- graph construction + stage-0 embed ----
    knn_kernel<<<BGR, 256, 0, stream>>>(pos, src);
    embed0_kernel<<<CEILDIV((long)NTOT*HID, 256), 256, 0, stream>>>(x, pos, st[0].We, st[0].be, Hb);

    float* h = Hb; float* t = Tb;
    int n = NTOT;
    for (int s = 0; s < 3; ++s) {
        const int d = DIMS[s];
        const int shift = s;
        const long nd = (long)n * d;

        // normalized-adjacency edge weights at this coarsening level
        fill0_kernel<<<CEILDIV(n, 256), 256, 0, stream>>>(deg, n);
        deg_kernel  <<<CEILDIV(nE, 256), 256, 0, stream>>>(src, shift, deg, nE);
        edgew_kernel<<<CEILDIV(nE, 256), 256, 0, stream>>>(src, deg, wE, shift, nE);

        for (int l = 0; l < NLAY; ++l) {
            const LayerP& L = st[s].L[l];
            // a = relu(bn(h)) -> bf16
            bn_stats_kernel<<<d, 256, 0, stream>>>(h, n, d, bnm, bnr);
            bnrelu_bf16_kernel<<<CEILDIV(nd, 256), 256, 0, stream>>>(h, bnm, bnr, L.bn_g, L.bn_b, A1, nd, d);
            // t1 = Lhat-gather(a)
            cheb_kernel<<<n, d, 0, stream>>>(A1, src, wE, shift, A2, d);
            // t = a@W0 + t1@W1 + cb   (fused dual-WMMA)
            transpose_cast_kernel<<<CEILDIV(d*d, 256), 256, 0, stream>>>(L.W0, WT1, d, d);
            transpose_cast_kernel<<<CEILDIV(d*d, 256), 256, 0, stream>>>(L.W1, WT2, d, d);
            launch_gemm(A1, A2, WT1, WT2, L.cb, t, n, d, d, stream);
            // a2 = relu(bn(t)) -> bf16 ; t = a2@mW + mb
            bn_stats_kernel<<<d, 256, 0, stream>>>(t, n, d, bnm, bnr);
            bnrelu_bf16_kernel<<<CEILDIV(nd, 256), 256, 0, stream>>>(t, bnm, bnr, L.m1g, L.m1b, A1, nd, d);
            transpose_cast_kernel<<<CEILDIV(d*d, 256), 256, 0, stream>>>(L.mW, WT1, d, d);
            launch_gemm(A1, nullptr, WT1, nullptr, L.mb, t, n, d, d, stream);
            // h = relu(bn(t)) + h   (residual, in place)
            bn_stats_kernel<<<d, 256, 0, stream>>>(t, n, d, bnm, bnr);
            bnrelu_add_kernel<<<CEILDIV(nd, 256), 256, 0, stream>>>(t, bnm, bnr, L.m2g, L.m2b, h, nd, d);
        }

        // pair pooling (graph coarsening)
        const int n2 = n / 2;
        pool_kernel<<<CEILDIV((long)n2*d, 256), 256, 0, stream>>>(h, t, (long)n2*d, d);
        { float* tmp = h; h = t; t = tmp; }
        n = n2;

        // next-stage embed GEMM (din=d, dout=2d) on WMMA
        if (s < 2) {
            const int d2 = DIMS[s+1];
            cast_bf16_kernel<<<CEILDIV((long)n*d, 256), 256, 0, stream>>>(h, A1, (long)n*d);
            transpose_cast_kernel<<<CEILDIV(d*d2, 256), 256, 0, stream>>>(st[s+1].We, WT1, d, d2);
            launch_gemm(A1, nullptr, WT1, nullptr, st[s+1].be, t, n, d, d2, stream);
            { float* tmp = h; h = t; t = tmp; }
        }
    }

    // ---- readout: graph mean (n=8192 -> 64x256) + MLP head + softmax ----
    graph_mean_kernel<<<64, 256, 0, stream>>>(h, G0);

    bn_stats_kernel<<<256, 256, 0, stream>>>(G0, 64, 256, bnm, bnr);
    bnrelu_f32_kernel<<<CEILDIV(64*256, 256), 256, 0, stream>>>(G0, bnm, bnr, hbn0g, hbn0b, G0, 64*256, 256);
    small_gemm_kernel<<<CEILDIV(64*128, 128), 128, 0, stream>>>(G0, hW0, hb0, G1, 64, 256, 128);

    bn_stats_kernel<<<128, 256, 0, stream>>>(G1, 64, 128, bnm, bnr);
    bnrelu_f32_kernel<<<CEILDIV(64*128, 256), 256, 0, stream>>>(G1, bnm, bnr, hbn1g, hbn1b, G1, 64*128, 128);
    small_gemm_kernel<<<CEILDIV(64*64, 128), 128, 0, stream>>>(G1, hW1, hb1, G2, 64, 128, 64);

    bn_stats_kernel<<<64, 256, 0, stream>>>(G2, 64, 64, bnm, bnr);
    bnrelu_f32_kernel<<<CEILDIV(64*64, 256), 256, 0, stream>>>(G2, bnm, bnr, hbn2g, hbn2b, G2, 64*64, 64);

    softmax_kernel<<<64, 64, 0, stream>>>(G2, (float*)d_out, 64);
}
